// ColBERT_49744311222945
// MI455X (gfx1250) — matrix operand order repaired and checked by reading, not compiled, not run on hardware
//
#include <hip/hip_runtime.h>
#include <hip/hip_bf16.h>

typedef __bf16 bf16;
typedef __attribute__((ext_vector_type(16))) __bf16 v16bf;
typedef __attribute__((ext_vector_type(8)))  __bf16 v8bf;
typedef __attribute__((ext_vector_type(8)))  float  v8f;

#define H_DIM   768
#define K_DIM   128
#define N_DOCS  2048
#define LD_TOK  128
#define B_Q     16
#define LQ_TOK  32
#define NEG_FILL (-100.0f)

union V16U { v16bf v; v8bf h[2]; };

static __device__ inline v8f splat8(float x) {
    v8f r;
#pragma unroll
    for (int i = 0; i < 8; ++i) r[i] = x;
    return r;
}

static __device__ inline v16bf load2x8(const bf16* p0, const bf16* p1) {
    V16U u;
    u.h[0] = *(const v8bf*)p0;
    u.h[1] = *(const v8bf*)p1;
    return u.v;
}

static __device__ inline v8f wmma_bf16(v16bf a, v16bf b, v8f c) {
    // D = A(16x32 bf16) * B(32x16 bf16) + C(16x16 f32)
    return __builtin_amdgcn_wmma_f32_16x16x32_bf16(
        /*neg_a=*/false, a, /*neg_b=*/false, b,
        /*c_mod=*/(short)0, c, /*reuse_a=*/false, /*reuse_b=*/false);
}

// ---------------------------------------------------------------------------
// Transpose + convert projection weight: W[H,K] f32 -> Wt[K,H] bf16
// ---------------------------------------------------------------------------
__global__ void wt_kernel(const float* __restrict__ W, bf16* __restrict__ Wt) {
    int idx = blockIdx.x * 256 + threadIdx.x;      // H*K = 98304 elements
    if (idx >= H_DIM * K_DIM) return;
    int k = idx % K_DIM;
    int h = idx / K_DIM;
    Wt[(size_t)k * H_DIM + h] = (bf16)W[(size_t)h * K_DIM + k];
}

// ---------------------------------------------------------------------------
// Projection + L2-normalize: enc[row, 0:128] = l2norm(hidden[row,:] @ W + b)
// Block = 256 threads (8 waves). Block handles 128 rows; wave w -> 16 rows.
// ---------------------------------------------------------------------------
__global__ void __launch_bounds__(256)
proj_norm_kernel(const float* __restrict__ hidden,   // [rows, 768]
                 const bf16*  __restrict__ Wt,       // [128, 768]
                 const float* __restrict__ bias,     // [128]
                 bf16*        __restrict__ enc)      // [rows, 128]
{
    const int wave  = threadIdx.x >> 5;
    const int lane  = threadIdx.x & 31;
    const int l15   = lane & 15;
    const int khalf = lane >> 4;
    const int row0  = blockIdx.x * 128 + wave * 16;

    v8f acc[8];
#pragma unroll
    for (int j = 0; j < 8; ++j) acc[j] = {};

    const float* arow = hidden + (size_t)(row0 + l15) * H_DIM;

    for (int h = 0; h < H_DIM; h += 32) {
        // --- A fragment: 16 rows x 32 k (fp32 -> bf16 on the fly) ---
        const float* p0 = arow + h + khalf * 8;        // K = khalf*8 .. +7
        const float* p1 = arow + h + 16 + khalf * 8;   // K = 16+khalf*8 .. +7
        float4 f0 = *(const float4*)(p0);
        float4 f1 = *(const float4*)(p0 + 4);
        float4 f2 = *(const float4*)(p1);
        float4 f3 = *(const float4*)(p1 + 4);
        v16bf a;
        a[0] = (bf16)f0.x; a[1] = (bf16)f0.y; a[2] = (bf16)f0.z; a[3] = (bf16)f0.w;
        a[4] = (bf16)f1.x; a[5] = (bf16)f1.y; a[6] = (bf16)f1.z; a[7] = (bf16)f1.w;
        a[8]  = (bf16)f2.x; a[9]  = (bf16)f2.y; a[10] = (bf16)f2.z; a[11] = (bf16)f2.w;
        a[12] = (bf16)f3.x; a[13] = (bf16)f3.y; a[14] = (bf16)f3.z; a[15] = (bf16)f3.w;

        // --- 8 N-tiles of W, each B frag: 32 k x 16 cols ---
#pragma unroll
        for (int j = 0; j < 8; ++j) {
            const bf16* bp = Wt + (size_t)(j * 16 + l15) * H_DIM + h + khalf * 8;
            v16bf b = load2x8(bp, bp + 16);
            acc[j] = wmma_bf16(a, b, acc[j]);
        }
    }

    // bias
#pragma unroll
    for (int j = 0; j < 8; ++j) {
        float bb = bias[j * 16 + l15];
#pragma unroll
        for (int i = 0; i < 8; ++i) acc[j][i] += bb;
    }

    // sum of squares per M-row (8 rows per lane-half), reduce across 16 lanes
    float ss[8];
#pragma unroll
    for (int i = 0; i < 8; ++i) {
        float s = 0.f;
#pragma unroll
        for (int j = 0; j < 8; ++j) s += acc[j][i] * acc[j][i];
        ss[i] = s;
    }
#pragma unroll
    for (int off = 1; off < 16; off <<= 1) {
#pragma unroll
        for (int i = 0; i < 8; ++i) ss[i] += __shfl_xor(ss[i], off, 16);
    }

    float scl[8];
#pragma unroll
    for (int i = 0; i < 8; ++i)
        scl[i] = 1.0f / fmaxf(sqrtf(ss[i]), 1e-12f);

    // store: lane owns col = j*16+l15, rows row0 + (khalf?8:0) + i
    const int rbase = row0 + (khalf ? 8 : 0);
#pragma unroll
    for (int j = 0; j < 8; ++j) {
        const int col = j * 16 + l15;
#pragma unroll
        for (int i = 0; i < 8; ++i)
            enc[(size_t)(rbase + i) * K_DIM + col] = (bf16)(acc[j][i] * scl[i]);
    }
}

// ---------------------------------------------------------------------------
// MaxSim: one wave per (b, n).  score[b,n] = sum_l qmask[l] *
//         max_m ( dmask[m] ? <Q[b,l], D[n,m]> : -100 )
// Block = 8 waves = 8 consecutive n for one b.
// ---------------------------------------------------------------------------
__global__ void __launch_bounds__(256)
maxsim_kernel(const bf16* __restrict__ Qenc,   // [B*LQ, 128]
              const bf16* __restrict__ Denc,   // [N*LD, 128]
              const int*  __restrict__ q_mask, // [B, LQ]
              const int*  __restrict__ d_mask, // [N, LD]
              float*      __restrict__ out)    // [B, N]
{
    const int wave  = threadIdx.x >> 5;
    const int lane  = threadIdx.x & 31;
    const int l15   = lane & 15;
    const int khalf = lane >> 4;
    const int n = blockIdx.x * 8 + wave;
    const int b = blockIdx.y;

    // Q fragments resident in registers: 2 M-tiles x 4 K-steps
    v16bf a[2][4];
#pragma unroll
    for (int mt = 0; mt < 2; ++mt) {
        const bf16* qrow = Qenc + (size_t)(b * LQ_TOK + mt * 16 + l15) * K_DIM;
#pragma unroll
        for (int kt = 0; kt < 4; ++kt) {
            const bf16* p = qrow + kt * 32 + khalf * 8;
            a[mt][kt] = load2x8(p, p + 16);
        }
    }

    v8f m0 = splat8(-3.402823466e38f);
    v8f m1 = splat8(-3.402823466e38f);

    const bf16* dbase = Denc + (size_t)n * LD_TOK * K_DIM;
    const int*  dm    = d_mask + (size_t)n * LD_TOK;

#pragma unroll
    for (int j = 0; j < 8; ++j) {
        const int tok = j * 16 + l15;
        const bf16* drow = dbase + (size_t)tok * K_DIM;
        if (j < 7) __builtin_prefetch(drow + 16 * K_DIM, 0, 1);

        v8f c0 = {};
        v8f c1 = {};
#pragma unroll
        for (int kt = 0; kt < 4; ++kt) {
            const bf16* p = drow + kt * 32 + khalf * 8;
            v16bf bfrag = load2x8(p, p + 16);
            c0 = wmma_bf16(a[0][kt], bfrag, c0);
            c1 = wmma_bf16(a[1][kt], bfrag, c1);
        }
        const bool ok = (dm[tok] != 0);
#pragma unroll
        for (int i = 0; i < 8; ++i) {
            float v0 = ok ? c0[i] : NEG_FILL;
            float v1 = ok ? c1[i] : NEG_FILL;
            m0[i] = fmaxf(m0[i], v0);
            m1[i] = fmaxf(m1[i], v1);
        }
    }

    // max over the 16 doc-token columns held by this half-wave
#pragma unroll
    for (int off = 1; off < 16; off <<= 1) {
#pragma unroll
        for (int i = 0; i < 8; ++i) {
            m0[i] = fmaxf(m0[i], __shfl_xor(m0[i], off, 16));
            m1[i] = fmaxf(m1[i], __shfl_xor(m1[i], off, 16));
        }
    }

    // apply q_mask, sum over query rows owned by this lane-half
    const int* qm = q_mask + (size_t)b * LQ_TOK;
    const int rbase = khalf ? 8 : 0;
    float s = 0.f;
#pragma unroll
    for (int i = 0; i < 8; ++i) {
        if (qm[rbase + i])      s += m0[i];   // rows 0..15   (M-tile 0)
        if (qm[16 + rbase + i]) s += m1[i];   // rows 16..31  (M-tile 1)
    }
    s += __shfl_xor(s, 16, 32);   // combine the two halves

    if (lane == 0) out[(size_t)b * N_DOCS + n] = s;
}

// ---------------------------------------------------------------------------
extern "C" void kernel_launch(void* const* d_in, const int* in_sizes, int n_in,
                              void* d_out, int out_size, void* d_ws, size_t ws_size,
                              hipStream_t stream) {
    const float* q_hidden = (const float*)d_in[0];   // [16,32,768]
    const int*   q_mask   = (const int*)  d_in[1];   // [16,32]
    const float* d_hidden = (const float*)d_in[2];   // [2048,128,768]
    const int*   d_mask   = (const int*)  d_in[3];   // [2048,128]
    const float* Wq       = (const float*)d_in[4];   // [768,128]
    const float* bq       = (const float*)d_in[5];   // [128]
    const float* Wd       = (const float*)d_in[6];   // [768,128]
    const float* bd       = (const float*)d_in[7];   // [128]
    float* out = (float*)d_out;                       // [16,2048]

    char* ws = (char*)d_ws;
    const size_t denc_bytes = (size_t)N_DOCS * LD_TOK * K_DIM * sizeof(bf16); // 64 MiB
    const size_t qenc_bytes = (size_t)B_Q * LQ_TOK * K_DIM * sizeof(bf16);    // 128 KiB
    const size_t wt_bytes   = (size_t)H_DIM * K_DIM * sizeof(bf16);           // 192 KiB
    bf16* Denc = (bf16*)(ws);
    bf16* Qenc = (bf16*)(ws + denc_bytes);
    bf16* Wdt  = (bf16*)(ws + denc_bytes + qenc_bytes);
    bf16* Wqt  = (bf16*)(ws + denc_bytes + qenc_bytes + wt_bytes);

    // 1) weight transpose + bf16 convert (tiny)
    wt_kernel<<<(H_DIM * K_DIM + 255) / 256, 256, 0, stream>>>(Wd, Wdt);
    wt_kernel<<<(H_DIM * K_DIM + 255) / 256, 256, 0, stream>>>(Wq, Wqt);

    // 2) projections + L2 normalize (D: memory-bound 805MB stream; Q: tiny)
    proj_norm_kernel<<<(N_DOCS * LD_TOK) / 128, 256, 0, stream>>>(d_hidden, Wdt, bd, Denc);
    proj_norm_kernel<<<(B_Q * LQ_TOK) / 128, 256, 0, stream>>>(q_hidden, Wqt, bq, Qenc);

    // 3) MaxSim scoring from L2-resident bf16 Denc
    maxsim_kernel<<<dim3(N_DOCS / 8, B_Q), 256, 0, stream>>>(Qenc, Denc, q_mask, d_mask, out);
}
